// Model_72438918414530
// MI455X (gfx1250) — matrix operand order repaired and checked by reading, not compiled, not run on hardware
//
#include <hip/hip_runtime.h>
#include <hip/hip_bf16.h>

// ---------------------------------------------------------------------------
// Sizes from the reference
// ---------------------------------------------------------------------------
#define BN   32      // batch
#define NN   512     // nodes
#define TT   64      // GRU timesteps
#define DD   128     // DLIT == DSEM == E
#define OO   64      // output embedding dim
#define NGTB 128     // 2 graphs x 2 types x 32 batch

typedef __attribute__((ext_vector_type(16))) _Float16 v16h;
typedef __attribute__((ext_vector_type(8)))  float    v8f;
typedef __attribute__((ext_vector_type(4)))  unsigned u32x4;
typedef __attribute__((ext_vector_type(8)))  int      i32x8;
typedef __attribute__((ext_vector_type(4)))  int      i32x4;

union HV16 { v16h h; unsigned u[8]; };

__device__ __forceinline__ v8f wmma16(v16h a, v16h b, v8f c) {
  return __builtin_amdgcn_wmma_f32_16x16x32_f16(false, a, false, b, (short)0, c,
                                                false, false);
}
__device__ __forceinline__ v8f vzero8() {
  v8f z;
#pragma unroll
  for (int i = 0; i < 8; ++i) z[i] = 0.0f;
  return z;
}
// A-operand (f16 16x32), contiguous K order chosen here (B matches): dword v of lane l holds K pair starting at:
__device__ __forceinline__ int a_kpair(int v, int lane) {
  return 2 * v + ((lane < 16) ? 0 : 16);
}
__device__ __forceinline__ float sigmoidf_(float x) { return 1.0f / (1.0f + __expf(-x)); }
__device__ __forceinline__ float lrelu_(float x) { return (x > 0.0f) ? x : 0.2f * x; }

// ---------------------------------------------------------------------------
// TDM: DMA one 32-row x 256B emb slab (contiguous 8KB) from global into LDS,
// padding +2 dwords after every 64 dwords -> LDS row stride = 132 halves.
// Descriptor per CDNA5 ISA ch.8 (group0: count/lds/global/type; group1: dims).
// ---------------------------------------------------------------------------
__device__ __forceinline__ void tdm_load_slab(const _Float16* gsrc, unsigned lds_addr) {
  unsigned long long ga = (unsigned long long)(size_t)gsrc;
  u32x4 g0;
  g0[0] = 1u;                                                 // count=1 (valid user D#)
  g0[1] = lds_addr;                                           // lds_addr [63:32]
  g0[2] = (unsigned)(ga & 0xFFFFFFFFu);                       // global_addr [95:64]
  g0[3] = (unsigned)((ga >> 32) & 0x1FFFFFFu) | (2u << 30);   // addr[56:32] | type=2
  i32x8 g1;
  g1[0] = (int)((2u << 16) | (1u << 20) | (5u << 22) | (1u << 25));
  //            data_size=4B | pad_en | pad_interval=64dw | pad_amount=2dw
  g1[1] = (int)(64u << 16);   // tensor_dim0 = 64 dwords (bits 79:48 low part)
  g1[2] = (int)(32u << 16);   // tensor_dim1 = 32 rows   (bits 111:80 low part)
  g1[3] = (int)(64u << 16);   // tile_dim0 = 64 dwords   (bits 127:112)
  g1[4] = 32;                 // tile_dim1 = 32 rows     (bits 143:128)
  g1[5] = 64;                 // tensor_dim0_stride = 64 dwords (bits 207:160 low)
  g1[6] = 0;
  g1[7] = 0;
  i32x4 gz;
#pragma unroll
  for (int i = 0; i < 4; ++i) gz[i] = 0;
#if defined(__clang_major__) && (__clang_major__ >= 23)
  i32x8 gz8;
#pragma unroll
  for (int i = 0; i < 8; ++i) gz8[i] = 0;
  __builtin_amdgcn_tensor_load_to_lds(g0, g1, gz, gz, gz8, 0);
#else
  __builtin_amdgcn_tensor_load_to_lds(g0, g1, gz, gz, 0);
#endif
}

// ---------------------------------------------------------------------------
// K0: build bit-packed adjacency masks + transposed f16 weight copies.
// ---------------------------------------------------------------------------
__global__ void prep_kernel(const int* __restrict__ CFG1, const int* __restrict__ LFG1,
                            const int* __restrict__ CFG2, const int* __restrict__ LFG2,
                            const float* __restrict__ Wl1, const float* __restrict__ Wl2,
                            const float* __restrict__ k1,  const float* __restrict__ k2,
                            const float* __restrict__ rk1, const float* __restrict__ rk2,
                            const float* __restrict__ m1,  const float* __restrict__ m2,
                            unsigned* __restrict__ Gmask,   // [128][512][16]
                            _Float16* __restrict__ wlitT,   // [2][128n][128k]
                            _Float16* __restrict__ mlp1T,   // [128n][128k]
                            _Float16* __restrict__ mlp2T,   // [128n][128k]
                            _Float16* __restrict__ kT,      // [2][384n][128k]
                            _Float16* __restrict__ rkT) {   // [2][384n][128k]
  const int NG = NGTB * NN * 16;   // 1048576
  const int NW = 2 * DD * DD;      // 32768
  const int NM = DD * DD;          // 16384
  const int NK = 2 * 3 * DD * DD;  // 98304
  const int total = NG + NW + 2 * NM + 2 * NK;
  for (int idx = blockIdx.x * blockDim.x + threadIdx.x; idx < total;
       idx += gridDim.x * blockDim.x) {
    if (idx < NG) {
      int ww = idx & 15, row = (idx >> 4) & (NN - 1), gtb = idx >> 13;
      int g = gtb >> 6, t = (gtb >> 5) & 1, b = gtb & 31;
      const int* G = g ? (t ? LFG2 : CFG2) : (t ? LFG1 : CFG1);
      const int* src = G + ((size_t)b * NN + row) * NN + ww * 32;
      unsigned m = 0u;
#pragma unroll
      for (int bit = 0; bit < 32; ++bit)
        if (src[bit]) m |= (1u << bit);
      Gmask[idx] = m;
    } else if (idx < NG + NW) {
      int r = idx - NG;
      int g = r >> 14, n = (r >> 7) & 127, k = r & 127;
      wlitT[r] = (_Float16)((g ? Wl2 : Wl1)[k * DD + n]);
    } else if (idx < NG + NW + NM) {
      int r = idx - (NG + NW);
      int n = r >> 7, k = r & 127;
      mlp1T[r] = (_Float16)(m1[k * DD + n]);
    } else if (idx < NG + NW + 2 * NM) {
      int r = idx - (NG + NW + NM);
      int n = r >> 7, k = r & 127;
      mlp2T[r] = (_Float16)(m2[k * DD + n]);
    } else if (idx < NG + NW + 2 * NM + NK) {
      int r = idx - (NG + NW + 2 * NM);
      int g = r / (3 * DD * DD), q = r % (3 * DD * DD);
      int n = q >> 7, k = q & 127;
      kT[r] = (_Float16)((g ? k2 : k1)[k * (3 * DD) + n]);
    } else {
      int r = idx - (NG + NW + 2 * NM + NK);
      int g = r / (3 * DD * DD), q = r % (3 * DD * DD);
      int n = q >> 7, k = q & 127;
      rkT[r] = (_Float16)((g ? rk2 : rk1)[k * (3 * DD) + n]);
    }
  }
}

// ---------------------------------------------------------------------------
// K1: lit = LITERAL @ W_lit (WMMA); also emb0 = relu(lit) in f16 for cfg+lfg.
// grid (8 rowblk, 32 b, 2 g), 256 threads (8 waves).
// ---------------------------------------------------------------------------
__global__ __launch_bounds__(256) void lit_kernel(const float* __restrict__ L1,
                                                  const float* __restrict__ L2,
                                                  const _Float16* __restrict__ wlitT,
                                                  float* __restrict__ lit_f32,
                                                  _Float16* __restrict__ embA) {
  int g = blockIdx.z, b = blockIdx.y, rowBase = blockIdx.x * 64;
  const float* LIT = g ? L2 : L1;
  const _Float16* WT = wlitT + (size_t)g * DD * DD;
  int lane = threadIdx.x & 31, w = threadIdx.x >> 5;
  int Mtile = w >> 1, n0 = (w & 1) * 64;
  v8f acc[4];
#pragma unroll
  for (int q = 0; q < 4; ++q) acc[q] = vzero8();
  int arow = rowBase + Mtile * 16 + (lane & 15);
  const float* ap = LIT + ((size_t)b * NN + arow) * DD;
  for (int ks = 0; ks < 4; ++ks) {
    v16h a;
#pragma unroll
    for (int v = 0; v < 8; ++v) {
      float2 p = *reinterpret_cast<const float2*>(ap + a_kpair(v, lane) + ks * 32);
      a[2 * v] = (_Float16)p.x;
      a[2 * v + 1] = (_Float16)p.y;
    }
#pragma unroll
    for (int nt = 0; nt < 4; ++nt) {
      int col = n0 + nt * 16 + (lane & 15);
      const _Float16* bp = WT + (size_t)col * DD + ks * 32 + ((lane < 16) ? 0 : 16);
      HV16 bv;
#pragma unroll
      for (int v = 0; v < 8; ++v) bv.u[v] = *reinterpret_cast<const unsigned*>(bp + 2 * v);
      acc[nt] = wmma16(a, bv.h, acc[nt]);
    }
  }
#pragma unroll
  for (int nt = 0; nt < 4; ++nt)
#pragma unroll
    for (int r = 0; r < 8; ++r) {
      int i2 = rowBase + Mtile * 16 + r + ((lane < 16) ? 0 : 8);
      int col = n0 + nt * 16 + (lane & 15);
      float val = acc[nt][r];
      lit_f32[((size_t)(g * BN + b) * NN + i2) * DD + col] = val;
      _Float16 rv = (_Float16)fmaxf(val, 0.0f);
      size_t e0 = ((size_t)((g * 2 + 0) * BN + b) * NN + i2) * DD + col;
      embA[e0] = rv;
      embA[e0 + (size_t)BN * NN * DD] = rv;  // same init for lfg slot
    }
}

// ---------------------------------------------------------------------------
// K2: xz = SEMANTIC @ k + bi for all timesteps. M=2048 rows per graph, N=384.
// grid (32 rowblk, 3 nblk, 2 g), 256 threads.
// ---------------------------------------------------------------------------
__global__ __launch_bounds__(256) void gru_xz_kernel(const float* __restrict__ S1,
                                                     const float* __restrict__ S2,
                                                     const _Float16* __restrict__ kT,
                                                     const float* __restrict__ gb1,
                                                     const float* __restrict__ gb2,
                                                     float* __restrict__ xz) {
  int g = blockIdx.z, nBase = blockIdx.y * 128, rowBase = blockIdx.x * 64;
  const float* SEM = g ? S2 : S1;
  const float* bi = g ? gb2 : gb1;  // row 0 of (2,384)
  const _Float16* KT = kT + (size_t)g * 3 * DD * DD;
  int lane = threadIdx.x & 31, w = threadIdx.x >> 5;
  int Mtile = w >> 1, n0 = (w & 1) * 64;
  v8f acc[4];
#pragma unroll
  for (int q = 0; q < 4; ++q) acc[q] = vzero8();
  int arow = rowBase + Mtile * 16 + (lane & 15);
  const float* ap = SEM + (size_t)arow * DD;
  for (int ks = 0; ks < 4; ++ks) {
    v16h a;
#pragma unroll
    for (int v = 0; v < 8; ++v) {
      float2 p = *reinterpret_cast<const float2*>(ap + a_kpair(v, lane) + ks * 32);
      a[2 * v] = (_Float16)p.x;
      a[2 * v + 1] = (_Float16)p.y;
    }
#pragma unroll
    for (int nt = 0; nt < 4; ++nt) {
      int col = nBase + n0 + nt * 16 + (lane & 15);
      const _Float16* bp = KT + (size_t)col * DD + ks * 32 + ((lane < 16) ? 0 : 16);
      HV16 bv;
#pragma unroll
      for (int v = 0; v < 8; ++v) bv.u[v] = *reinterpret_cast<const unsigned*>(bp + 2 * v);
      acc[nt] = wmma16(a, bv.h, acc[nt]);
    }
  }
#pragma unroll
  for (int nt = 0; nt < 4; ++nt)
#pragma unroll
    for (int r = 0; r < 8; ++r) {
      int i2 = rowBase + Mtile * 16 + r + ((lane < 16) ? 0 : 8);
      int col = nBase + n0 + nt * 16 + (lane & 15);
      xz[(size_t)g * 2048 * 384 + (size_t)i2 * 384 + col] = acc[nt][r] + bi[col];
    }
}

// ---------------------------------------------------------------------------
// K3: GRU recurrence, 64 steps. One WG (512 thr, 16 waves) per graph.
// h kept in registers (8 elems/thread) + f16 LDS mirror for WMMA A operand.
// ---------------------------------------------------------------------------
__global__ __launch_bounds__(512) void gru_rec_kernel(const _Float16* __restrict__ rkT,
                                                      const float* __restrict__ gb1,
                                                      const float* __restrict__ gb2,
                                                      const float* __restrict__ xz,
                                                      float* __restrict__ sem) {
  __shared__ float s_hz[BN][392];          // 50176 B
  __shared__ _Float16 s_h16[BN][136];      //  8704 B
  int g = blockIdx.x;
  const _Float16* RKT = rkT + (size_t)g * 3 * DD * DD;
  const float* br = (g ? gb2 : gb1) + 384;  // row 1 of (2,384)
  const float* XZ = xz + (size_t)g * 2048 * 384;
  int tid = threadIdx.x, lane = tid & 31, w = tid >> 5;
  float hreg[8];
#pragma unroll
  for (int s = 0; s < 8; ++s) {
    hreg[s] = 0.0f;
    int p = tid + s * 512;
    s_h16[p >> 7][p & 127] = (_Float16)0.0f;
  }
  __syncthreads();
  for (int t = 0; t < TT; ++t) {
    // hz = h @ rk  (M=32, N=384)  -> 48 tiles / 16 waves = 3 each
#pragma unroll
    for (int q = 0; q < 3; ++q) {
      int id = w * 3 + q;
      int Mtile = id & 1, Ntile = id >> 1;
      v8f acc = vzero8();
      int arow = Mtile * 16 + (lane & 15);
      int col = Ntile * 16 + (lane & 15);
#pragma unroll
      for (int ks = 0; ks < 4; ++ks) {
        HV16 av;
#pragma unroll
        for (int v = 0; v < 8; ++v)
          av.u[v] = *reinterpret_cast<const unsigned*>(&s_h16[arow][a_kpair(v, lane) + ks * 32]);
        const _Float16* bp = RKT + (size_t)col * DD + ks * 32 + ((lane < 16) ? 0 : 16);
        HV16 bv;
#pragma unroll
        for (int v = 0; v < 8; ++v) bv.u[v] = *reinterpret_cast<const unsigned*>(bp + 2 * v);
        acc = wmma16(av.h, bv.h, acc);
      }
#pragma unroll
      for (int r = 0; r < 8; ++r) {
        int M = Mtile * 16 + r + ((lane < 16) ? 0 : 8);
        s_hz[M][col] = acc[r] + br[col];
      }
    }
    __syncthreads();
    // gates
#pragma unroll
    for (int s = 0; s < 8; ++s) {
      int p = tid + s * 512;
      int m = p >> 7, n = p & 127;
      const float* xr = XZ + (size_t)(m * TT + t) * 384;
      float z = sigmoidf_(xr[n] + s_hz[m][n]);
      float r_ = sigmoidf_(xr[128 + n] + s_hz[m][128 + n]);
      float hh = tanhf(xr[256 + n] + r_ * s_hz[m][256 + n]);
      float hn = z * hreg[s] + (1.0f - z) * hh;
      hreg[s] = hn;
      s_h16[m][n] = (_Float16)hn;
    }
    __syncthreads();
  }
#pragma unroll
  for (int s = 0; s < 8; ++s) {
    int p = tid + s * 512;
    sem[(size_t)g * BN * DD + p] = hreg[s];
  }
}

// ---------------------------------------------------------------------------
// K4: one GNN iteration (attention softmax + A@emb + 2-layer MLP + tanh(lit+.)).
// grid (8 rowblk, 128 gtb), 256 threads (8 waves). All matmuls via WMMA f16.
// emb K-slabs double-buffered in LDS via TDM tensor_load_to_lds (TENSORcnt).
// ---------------------------------------------------------------------------
__global__ __launch_bounds__(256) void iter_kernel(
    const _Float16* __restrict__ emb_in, _Float16* __restrict__ emb_out,
    const unsigned* __restrict__ Gmask, const float* __restrict__ lit_f32,
    const _Float16* __restrict__ mlp1T, const _Float16* __restrict__ mlp2T,
    const float* __restrict__ c1w, const float* __restrict__ c1b,
    const float* __restrict__ c2w, const float* __restrict__ c2b,
    const float* __restrict__ l1w, const float* __restrict__ l1b,
    const float* __restrict__ l2w, const float* __restrict__ l2b) {
  __shared__ float s_ah1[NN];
  __shared__ float s_ah2[NN];
  __shared__ float s_rowm[64];
  __shared__ float s_rowinv[64];
  // phase c: two TDM slab buffers [32][132] f16 @ offsets 0, 8704
  // phase d/e: same region reused as tmp2 [64][136] f16 (17408 B)
  __shared__ __align__(16) unsigned char s_mix[17408];
  __shared__ __align__(16) _Float16 s_tmp[64][136];
  _Float16(*tmp2)[136] = reinterpret_cast<_Float16(*)[136]>(s_mix);

  int gtb = blockIdx.y;
  int g = gtb >> 6, t = (gtb >> 5) & 1, b = gtb & 31;
  int rowBase = blockIdx.x * 64;
  const _Float16* EIN = emb_in + (size_t)gtb * NN * DD;
  _Float16* EOUT = emb_out + (size_t)gtb * NN * DD;
  const unsigned* GM = Gmask + (size_t)gtb * NN * 16;
  const float* LITR = lit_f32 + (size_t)(g * BN + b) * NN * DD;
  const float* w1 = t ? l1w : c1w;
  const float* w2 = t ? l2w : c2w;
  float b1 = (t ? l1b : c1b)[0];
  float b2 = (t ? l2b : c2b)[0];
  int tid = threadIdx.x, lane = tid & 31, w = tid >> 5;

  // ---- phase a: ah1/ah2 for all 512 nodes ----
  for (int rep = 0; rep < 2; ++rep) {
    int j = tid + rep * 256;
    const _Float16* er = EIN + (size_t)j * DD;
    float s1 = 0.0f, s2 = 0.0f;
    for (int k = 0; k < DD; ++k) {
      float e = (float)er[k];
      s1 += e * w1[k];
      s2 += e * w2[k];
    }
    s_ah1[j] = s1 + b1;
    s_ah2[j] = s2 + b2;
  }
  __syncthreads();

  // ---- phase b: per-row masked max + exp-sum (8 rows per wave) ----
  for (int rr = 0; rr < 8; ++rr) {
    int lrow = w * 8 + rr;
    int grow = rowBase + lrow;
    float ah1v = s_ah1[grow];
    const unsigned* gr = GM + (size_t)grow * 16;
    unsigned Gw[16];
    float vals[16];
#pragma unroll
    for (int c = 0; c < 16; ++c) Gw[c] = gr[c];
    float m = -3.0e38f;
#pragma unroll
    for (int c = 0; c < 16; ++c) {
      float v = lrelu_(ah1v + s_ah2[c * 32 + lane]);
      vals[c] = v;
      if ((Gw[c] >> lane) & 1u) m = fmaxf(m, v);
    }
#pragma unroll
    for (int off = 16; off; off >>= 1) m = fmaxf(m, __shfl_xor(m, off, 32));
    float ssum = 0.0f;
#pragma unroll
    for (int c = 0; c < 16; ++c)
      if ((Gw[c] >> lane) & 1u) ssum += __expf(vals[c] - m);
#pragma unroll
    for (int off = 16; off; off >>= 1) ssum += __shfl_xor(ssum, off, 32);
    if (lane == 0) {
      s_rowm[lrow] = m;
      s_rowinv[lrow] = (ssum > 0.0f) ? 1.0f / ssum : 0.0f;
    }
  }
  __syncthreads();

  // ---- phase c: tmp = A @ emb_in  (M=64, N=128, K=512), TDM double-buffered ----
  int Mtile = w >> 1, n0 = (w & 1) * 64;
  int arow_l = Mtile * 16 + (lane & 15);
  int arow_g = rowBase + arow_l;
  float ah1v = s_ah1[arow_g];
  float rm = s_rowm[arow_l], rinv = s_rowinv[arow_l];
  const unsigned* grow = GM + (size_t)arow_g * 16;
  unsigned slab_base = (unsigned)(size_t)(&s_mix[0]);
  v8f acc[4];
#pragma unroll
  for (int q = 0; q < 4; ++q) acc[q] = vzero8();
  if (w == 0) tdm_load_slab(EIN, slab_base);  // prime slab 0 (ks=0)
  for (int ks = 0; ks < 16; ++ks) {
    int p = ks & 1;
    if (w == 0) {
      if (ks + 1 < 16) {
        tdm_load_slab(EIN + (size_t)(ks + 1) * 32 * DD, slab_base + (unsigned)((p ^ 1) * 8704));
        __builtin_amdgcn_s_wait_tensorcnt(1);  // slab ks done, ks+1 in flight
      } else {
        __builtin_amdgcn_s_wait_tensorcnt(0);
      }
    }
    __syncthreads();  // slab p ready for all waves
    const _Float16(*slab)[132] = reinterpret_cast<const _Float16(*)[132]>(s_mix + p * 8704);
    unsigned Gw = grow[ks];
    v16h a;
#pragma unroll
    for (int v = 0; v < 8; ++v) {
      int kk0 = a_kpair(v, lane);
#pragma unroll
      for (int s = 0; s < 2; ++s) {
        int K = kk0 + s;
        float av = 0.0f;
        if ((Gw >> K) & 1u) {
          float x = lrelu_(ah1v + s_ah2[ks * 32 + K]);
          av = __expf(x - rm) * rinv;
        }
        a[2 * v + s] = (_Float16)av;
      }
    }
    int kb2 = (lane < 16) ? 0 : 16;
#pragma unroll
    for (int nt = 0; nt < 4; ++nt) {
      int col = n0 + nt * 16 + (lane & 15);
      HV16 bv;
#pragma unroll
      for (int v = 0; v < 8; ++v) {
        unsigned lo = *reinterpret_cast<const unsigned short*>(&slab[kb2 + 2 * v][col]);
        unsigned hi = *reinterpret_cast<const unsigned short*>(&slab[kb2 + 2 * v + 1][col]);
        bv.u[v] = lo | (hi << 16);
      }
      acc[nt] = wmma16(a, bv.h, acc[nt]);
    }
    __syncthreads();  // all reads of slab p done before TDM re-targets it
  }
#pragma unroll
  for (int nt = 0; nt < 4; ++nt)
#pragma unroll
    for (int r = 0; r < 8; ++r) {
      int Ml = Mtile * 16 + r + ((lane < 16) ? 0 : 8);
      s_tmp[Ml][n0 + nt * 16 + (lane & 15)] = (_Float16)acc[nt][r];
    }
  __syncthreads();

  // ---- phase d: tmp2 = relu(tmp @ mlp1) ----
  v8f acc2[4];
#pragma unroll
  for (int q = 0; q < 4; ++q) acc2[q] = vzero8();
#pragma unroll
  for (int ks = 0; ks < 4; ++ks) {
    HV16 av;
#pragma unroll
    for (int v = 0; v < 8; ++v)
      av.u[v] = *reinterpret_cast<const unsigned*>(&s_tmp[arow_l][a_kpair(v, lane) + ks * 32]);
#pragma unroll
    for (int nt = 0; nt < 4; ++nt) {
      int col = n0 + nt * 16 + (lane & 15);
      const _Float16* bp = mlp1T + (size_t)col * DD + ks * 32 + ((lane < 16) ? 0 : 16);
      HV16 bv;
#pragma unroll
      for (int v = 0; v < 8; ++v) bv.u[v] = *reinterpret_cast<const unsigned*>(bp + 2 * v);
      acc2[nt] = wmma16(av.h, bv.h, acc2[nt]);
    }
  }
  __syncthreads();  // slab region reuse as tmp2
#pragma unroll
  for (int nt = 0; nt < 4; ++nt)
#pragma unroll
    for (int r = 0; r < 8; ++r) {
      int Ml = Mtile * 16 + r + ((lane < 16) ? 0 : 8);
      tmp2[Ml][n0 + nt * 16 + (lane & 15)] = (_Float16)fmaxf(acc2[nt][r], 0.0f);
    }
  __syncthreads();

  // ---- phase e: emb_out = tanh(lit + tmp2 @ mlp2) ----
  v8f acc3[4];
#pragma unroll
  for (int q = 0; q < 4; ++q) acc3[q] = vzero8();
#pragma unroll
  for (int ks = 0; ks < 4; ++ks) {
    HV16 av;
#pragma unroll
    for (int v = 0; v < 8; ++v)
      av.u[v] = *reinterpret_cast<const unsigned*>(&tmp2[arow_l][a_kpair(v, lane) + ks * 32]);
#pragma unroll
    for (int nt = 0; nt < 4; ++nt) {
      int col = n0 + nt * 16 + (lane & 15);
      const _Float16* bp = mlp2T + (size_t)col * DD + ks * 32 + ((lane < 16) ? 0 : 16);
      HV16 bv;
#pragma unroll
      for (int v = 0; v < 8; ++v) bv.u[v] = *reinterpret_cast<const unsigned*>(bp + 2 * v);
      acc3[nt] = wmma16(av.h, bv.h, acc3[nt]);
    }
  }
#pragma unroll
  for (int nt = 0; nt < 4; ++nt)
#pragma unroll
    for (int r = 0; r < 8; ++r) {
      int Ml = Mtile * 16 + r + ((lane < 16) ? 0 : 8);
      int grow2 = rowBase + Ml;
      int col = n0 + nt * 16 + (lane & 15);
      float lv = LITR[(size_t)grow2 * DD + col];
      EOUT[(size_t)grow2 * DD + col] = (_Float16)tanhf(lv + acc3[nt][r]);
    }
}

// ---------------------------------------------------------------------------
// K5: mid = [sum_i cfg_e, sum_i lfg_e, sem];  outemb = mid @ Wout + bout.
// grid (32 b, 2 g), 128 threads.
// ---------------------------------------------------------------------------
__global__ __launch_bounds__(128) void reduce_kernel(const _Float16* __restrict__ embF,
                                                     const float* __restrict__ sem,
                                                     const float* __restrict__ Wout,
                                                     const float* __restrict__ bout,
                                                     float* __restrict__ outemb) {
  __shared__ float mid[3 * DD];
  int b = blockIdx.x, g = blockIdx.y, tid = threadIdx.x;
  {
    const _Float16* e0 = embF + ((size_t)((g * 2 + 0) * BN + b) * NN) * DD + tid;
    const _Float16* e1 = embF + ((size_t)((g * 2 + 1) * BN + b) * NN) * DD + tid;
    float s0 = 0.0f, s1 = 0.0f;
    for (int i = 0; i < NN; ++i) {
      s0 += (float)e0[(size_t)i * DD];
      s1 += (float)e1[(size_t)i * DD];
    }
    mid[tid] = s0;
    mid[DD + tid] = s1;
    mid[2 * DD + tid] = sem[(size_t)g * BN * DD + b * DD + tid];
  }
  __syncthreads();
  if (tid < OO) {
    float s = bout[tid];
    for (int k = 0; k < 3 * DD; ++k) s += mid[k] * Wout[k * OO + tid];
    outemb[(size_t)(g * BN + b) * OO + tid] = s;
  }
}

// ---------------------------------------------------------------------------
// K6: cosine similarity -> (dot+1)/2.
// ---------------------------------------------------------------------------
__global__ void final_kernel(const float* __restrict__ outemb, float* __restrict__ out) {
  int b = threadIdx.x;
  if (b >= BN) return;
  const float* e1 = outemb + (size_t)b * OO;
  const float* e2 = outemb + (size_t)(BN + b) * OO;
  float d = 0.0f, n1 = 0.0f, n2 = 0.0f;
  for (int o = 0; o < OO; ++o) {
    float a = e1[o], c = e2[o];
    d += a * c;
    n1 += a * a;
    n2 += c * c;
  }
  float cs = d * rsqrtf(fmaxf(n1, 1e-12f)) * rsqrtf(fmaxf(n2, 1e-12f));
  out[b] = (cs + 1.0f) * 0.5f;
}

// ---------------------------------------------------------------------------
// Host launcher
// ---------------------------------------------------------------------------
extern "C" void kernel_launch(void* const* d_in, const int* in_sizes, int n_in,
                              void* d_out, int out_size, void* d_ws, size_t ws_size,
                              hipStream_t stream) {
  const int* CFG1 = (const int*)d_in[0];
  const int* LFG1 = (const int*)d_in[1];
  const float* LIT1 = (const float*)d_in[2];
  const float* SEM1 = (const float*)d_in[3];
  const int* CFG2 = (const int*)d_in[4];
  const int* LFG2 = (const int*)d_in[5];
  const float* LIT2 = (const float*)d_in[6];
  const float* SEM2 = (const float*)d_in[7];
  const float* Wl1 = (const float*)d_in[8];
  const float* k1 = (const float*)d_in[9];
  const float* rk1 = (const float*)d_in[10];
  const float* gb1 = (const float*)d_in[11];
  const float* Wl2 = (const float*)d_in[12];
  const float* k2 = (const float*)d_in[13];
  const float* rk2 = (const float*)d_in[14];
  const float* gb2 = (const float*)d_in[15];
  const float* c1w = (const float*)d_in[16];
  const float* c1b = (const float*)d_in[17];
  const float* c2w = (const float*)d_in[18];
  const float* c2b = (const float*)d_in[19];
  const float* l1w = (const float*)d_in[20];
  const float* l1b = (const float*)d_in[21];
  const float* l2w = (const float*)d_in[22];
  const float* l2b = (const float*)d_in[23];
  const float* m1 = (const float*)d_in[24];
  const float* m2 = (const float*)d_in[25];
  const float* Wout = (const float*)d_in[26];
  const float* bout = (const float*)d_in[27];

  // workspace carve-out (~61 MB total)
  char* ws = (char*)d_ws;
  size_t off = 0;
  auto carve = [&](size_t bytes) {
    void* p = ws + off;
    off = (off + bytes + 255) & ~(size_t)255;
    return p;
  };
  float* lit = (float*)carve((size_t)2 * BN * NN * DD * 4);           // 16 MB
  _Float16* embA = (_Float16*)carve((size_t)NGTB * NN * DD * 2);      // 16 MB
  _Float16* embB = (_Float16*)carve((size_t)NGTB * NN * DD * 2);      // 16 MB
  unsigned* Gm = (unsigned*)carve((size_t)NGTB * NN * 16 * 4);        // 4 MB
  float* xz = (float*)carve((size_t)2 * 2048 * 384 * 4);              // 6.3 MB
  float* sem = (float*)carve((size_t)2 * BN * DD * 4);
  float* outemb = (float*)carve((size_t)2 * BN * OO * 4);
  _Float16* wlitT = (_Float16*)carve((size_t)2 * DD * DD * 2);
  _Float16* mlp1T = (_Float16*)carve((size_t)DD * DD * 2);
  _Float16* mlp2T = (_Float16*)carve((size_t)DD * DD * 2);
  _Float16* kT = (_Float16*)carve((size_t)2 * 3 * DD * DD * 2);
  _Float16* rkT = (_Float16*)carve((size_t)2 * 3 * DD * DD * 2);
  (void)in_sizes; (void)n_in; (void)out_size; (void)ws_size;

  prep_kernel<<<2048, 256, 0, stream>>>(CFG1, LFG1, CFG2, LFG2, Wl1, Wl2, k1, k2,
                                        rk1, rk2, m1, m2, Gm, wlitT, mlp1T, mlp2T, kT, rkT);
  lit_kernel<<<dim3(8, BN, 2), 256, 0, stream>>>(LIT1, LIT2, wlitT, lit, embA);
  gru_xz_kernel<<<dim3(32, 3, 2), 256, 0, stream>>>(SEM1, SEM2, kT, gb1, gb2, xz);
  gru_rec_kernel<<<2, 512, 0, stream>>>(rkT, gb1, gb2, xz, sem);

  _Float16* ein = embA;
  _Float16* eout = embB;
  for (int it = 0; it < 5; ++it) {
    iter_kernel<<<dim3(8, NGTB), 256, 0, stream>>>(ein, eout, Gm, lit, mlp1T, mlp2T,
                                                   c1w, c1b, c2w, c2b, l1w, l1b, l2w, l2b);
    _Float16* tswap = ein; ein = eout; eout = tswap;
  }
  reduce_kernel<<<dim3(BN, 2), 128, 0, stream>>>(ein, sem, Wout, bout, outemb);
  final_kernel<<<1, 64, 0, stream>>>(outemb, (float*)d_out);
}
